// MultiheadAttentionRelative_49495203119592
// MI455X (gfx1250) — matrix operand order repaired and checked by reading, not compiled, not run on hardware
//
#include <hip/hip_runtime.h>
#include <hip/hip_bf16.h>
#include <stdint.h>

// ---------------------------------------------------------------------------
// MI455X (gfx1250) relative-position multi-head attention.
//   S=1024, B=4, E=1024, H=16, hd=64, table=2047
// All GEMM work on v_wmma_f32_16x16x32_bf16 (f32 accumulate).
// All operands live in "fragment-ready" packed layouts: each lane's 16 bf16
// elements contiguous (32B) -> every fragment load is 2x b128.
// Projection GEMMs are register-blocked 2xM so each B-fragment feeds 2 wmmas.
// ---------------------------------------------------------------------------

typedef __attribute__((ext_vector_type(16))) __bf16 v16bf;
typedef __attribute__((ext_vector_type(8)))  float  v8f;

union B16x16 { v16bf v; unsigned short u[16]; uint4 q[2]; };

#define S_LEN 1024
#define BATCH 4
#define EMB   1024
#define HEADS 16
#define HD    64
#define TBL_CT 128          // packed table: 128 column-tiles of 16 (covers 0..2047)
#define MROWS (S_LEN * BATCH)

static __device__ __forceinline__ unsigned short f32_to_bf16(float f) {
  unsigned int u = __float_as_uint(f);
  unsigned int r = u + 0x7FFFu + ((u >> 16) & 1u);   // RNE
  return (unsigned short)(r >> 16);
}
static __device__ __forceinline__ float bf16_to_f32(unsigned short h) {
  return __uint_as_float(((unsigned int)h) << 16);
}

static __device__ __forceinline__ v8f wmma_bf16(const B16x16& a, const B16x16& b, v8f c) {
  return __builtin_amdgcn_wmma_f32_16x16x32_bf16(false, a.v, false, b.v,
                                                 (short)0, c, false, false);
}

// A-fragment element index for K value k (16x32 bf16 A, this code's contiguous k order):
//   lane = (k>>4)*16 + m ; element e = k & 15
static __device__ __forceinline__ int a_e(int k) {
  return k & 15;
}
static __device__ __forceinline__ int a_lane(int k, int m) {
  return ((k >> 4) << 4) | m;
}
// B-fragment (32x16): lane = (k>>4)*16 + n ; element e = k & 15

// One fragment = 32 contiguous bytes per lane.
static __device__ __forceinline__ void load_frag(B16x16& F, const unsigned short* p) {
  const uint4* s = (const uint4*)p;
  uint4 x = s[0], y = s[1];
  F.q[0] = x; F.q[1] = y;
}

// ---------------------------------------------------------------------------
// Pack E x E weight (f32, W[n][k]) into B-fragment-ready bf16 tiles:
//   PW[((kt*64 + nt)*32 + lane)*16 + e] = bf16(W[nt*16 + (lane&15)][kt*32 + 16*(lane>>4) + e])
// ---------------------------------------------------------------------------
__global__ __launch_bounds__(256) void k_pack_w(const float* __restrict__ W,
                                                unsigned short* __restrict__ PW) {
  const int i = blockIdx.x * blockDim.x + threadIdx.x;   // grid covers EMB*EMB
  const int e    = i & 15;
  const int lane = (i >> 4) & 31;
  const int nt   = (i >> 9) & 63;
  const int kt   = i >> 15;
  const int n = nt * 16 + (lane & 15);
  const int k = kt * 32 + ((lane >> 4) << 4) + e;
  PW[i] = f32_to_bf16(W[(size_t)n * EMB + k]);
}

// Pack rel_pos_table (f32 [2047][64]) into B-fragment-ready bf16 tiles:
//   PT[((dt*128 + ct)*32 + lane)*16 + e], k-dim = d, n-dim = table column c.
__global__ __launch_bounds__(256) void k_pack_tab(const float* __restrict__ tab,
                                                  unsigned short* __restrict__ PT) {
  const int i = blockIdx.x * blockDim.x + threadIdx.x;   // 2*128*512 = 131072
  if (i >= 2 * TBL_CT * 512) return;
  const int e    = i & 15;
  const int lane = (i >> 4) & 31;
  const int ct   = (i >> 9) & (TBL_CT - 1);
  const int dt   = i >> 16;
  int c = ct * 16 + (lane & 15);
  c = c > 2044 ? 2044 : c;                               // clip(dist)+half-1 max
  const int d = dt * 32 + ((lane >> 4) << 4) + e;
  PT[i] = f32_to_bf16(tab[(size_t)c * HD + d]);
}

// ---------------------------------------------------------------------------
// Fused QKV projection. Block 256 (8 waves), tile 128(M) x 128(N), K-step 32.
// Each wave: 2 m-tiles x 4 n-tiles = 8 accumulators; every B-frag -> 2 wmmas.
// A-tile: LDS, A-fragment-ready, double buffered. B: packed global.
// Outputs scattered into fragment-ready packed layouts:
//   mode 0: Q (scaled) -> PQ  A-ready tiles [bh][st(64)][dt(2)][lane][16]
//   mode 1: K          -> PK  B-ready tiles [bh][tt(64)][dt(2)][lane][16]  (k-dim=d)
//   mode 2: V          -> PV  B-ready tiles [bh][tc(32)][nd(4)][lane][16]  (k-dim=t)
// ---------------------------------------------------------------------------
__global__ __launch_bounds__(256) void k_qkv(
    const float* __restrict__ Xq, const float* __restrict__ Xk, const float* __restrict__ Xv,
    const unsigned short* __restrict__ PWq, const unsigned short* __restrict__ PWk,
    const unsigned short* __restrict__ PWv,
    const float* __restrict__ bq, const float* __restrict__ bk, const float* __restrict__ bv,
    unsigned short* __restrict__ PQ, unsigned short* __restrict__ PK,
    unsigned short* __restrict__ PV, float scale) {
  __shared__ unsigned short Xs[2][8][512];   // [buf][m-tile][lane*16+e], A-frag-ready

  const int mode = blockIdx.z;
  const float* __restrict__ X = (mode == 0) ? Xq : (mode == 1) ? Xk : Xv;
  const unsigned short* __restrict__ PW = (mode == 0) ? PWq : (mode == 1) ? PWk : PWv;
  const float* __restrict__ bias = (mode == 0) ? bq : (mode == 1) ? bk : bv;

  const int m0 = blockIdx.x * 128;
  const int n0 = blockIdx.y * 128;
  const int tid = threadIdx.x;
  const int w = tid >> 5, lane = tid & 31;
  const int half = lane >> 4, n15 = lane & 15;
  const int wm = w & 3, wn = w >> 2;

  auto stage = [&](int kt, int buf) {
    const int r = tid >> 1, c0 = (tid & 1) * 16;   // 128 rows x 32 cols
    const float* src = &X[(size_t)(m0 + r) * EMB + kt * 32 + c0];
    const int m = r & 15, mt = r >> 4;
#pragma unroll
    for (int j = 0; j < 16; ++j) {
      const int k = c0 + j;
      Xs[buf][mt][a_lane(k, m) * 16 + a_e(k)] = f32_to_bf16(src[j]);
    }
  };

  v8f acc[2][4] = {};
  stage(0, 0);
  const int ntg0 = (n0 >> 4) + wn * 4;

  for (int kt = 0; kt < 32; ++kt) {
    __syncthreads();
    if (kt + 1 < 32) stage(kt + 1, (kt + 1) & 1);
    B16x16 a0, a1;
    load_frag(a0, &Xs[kt & 1][wm * 2 + 0][lane * 16]);
    load_frag(a1, &Xs[kt & 1][wm * 2 + 1][lane * 16]);
#pragma unroll
    for (int a = 0; a < 4; ++a) {
      B16x16 bfr;
      load_frag(bfr, &PW[((size_t)(kt * 64 + ntg0 + a) * 32 + lane) * 16]);
      acc[0][a] = wmma_bf16(a0, bfr, acc[0][a]);
      acc[1][a] = wmma_bf16(a1, bfr, acc[1][a]);   // B-frag reused
    }
  }

  // Epilogue: bias (+scale for Q), scatter bf16 into packed layouts.
#pragma unroll
  for (int mi = 0; mi < 2; ++mi) {
#pragma unroll
    for (int a = 0; a < 4; ++a) {
      const int ng = n0 + wn * 64 + a * 16 + n15;
      const float bval = bias[ng];
      const int h = ng >> 6, d = ng & 63;
#pragma unroll
      for (int r = 0; r < 8; ++r) {
        const int mg = m0 + (wm * 2 + mi) * 16 + r + 8 * half;   // row = s*B + b
        float val = acc[mi][a][r] + bval;
        if (mode == 0) val *= scale;
        const unsigned short o16 = f32_to_bf16(val);
        const int s = mg >> 2, b = mg & 3;
        const int bh = b * HEADS + h;
        if (mode == 0) {
          const int st = s >> 4, m = s & 15, dt = d >> 5, kin = d & 31;
          PQ[(((size_t)(bh * 64 + st) * 2 + dt) * 32 + a_lane(kin, m)) * 16 + a_e(kin)] = o16;
        } else if (mode == 1) {
          const int tt = s >> 4, dt = d >> 5, kin = d & 31;      // B: k=d, n=t
          PK[(((size_t)(bh * 64 + tt) * 2 + dt) * 32 + (kin >> 4) * 16 + (s & 15)) * 16 + (kin & 15)] = o16;
        } else {
          const int tc = s >> 5, nd = d >> 4, kin = s & 31;      // B: k=t, n=d
          PV[(((size_t)(bh * 32 + tc) * 4 + nd) * 32 + (kin >> 4) * 16 + (d & 15)) * 16 + (kin & 15)] = o16;
        }
      }
    }
  }
}

// ---------------------------------------------------------------------------
// Flash attention. grid=(B*H, S/64), block=128 (4 independent waves; wave w
// owns 16 s-rows). Per 32-wide t-chunk:
//   content scores : 4 wmma (PK frags, global b128)
//   rel-pos band   : 8 wmma (PT frags) -> LDS, gathered as P[s, t-s+1022]
//   online softmax : f32, shfl_xor within 16-lane halves
//   O += probs x V : 4 wmma (probs relayout via LDS b128)
// No cross-wave sharing -> no block barriers in the loop.
// ---------------------------------------------------------------------------
__global__ __launch_bounds__(128) void k_attn(
    const unsigned short* __restrict__ PQ, const unsigned short* __restrict__ PK,
    const unsigned short* __restrict__ PV, const unsigned short* __restrict__ PT,
    unsigned short* __restrict__ PO) {
  __shared__ unsigned short pos_s[4][16 * 64];   // per-wave pos band (C layout)
  __shared__ unsigned short pr_s[4][512];        // per-wave probs (A-frag-ready)

  const int bh = blockIdx.x;
  const int b = bh >> 4, h = bh & 15;
  const int s0 = blockIdx.y * 64;
  const int tid = threadIdx.x;
  const int w = tid >> 5, lane = tid & 31;
  const int half = lane >> 4, n15 = lane & 15;
  const int sw0 = w * 16;

  B16x16 qa[2];
  {
    const int st = (s0 >> 4) + w;
    load_frag(qa[0], &PQ[(((size_t)(bh * 64 + st) * 2 + 0) * 32 + lane) * 16]);
    load_frag(qa[1], &PQ[(((size_t)(bh * 64 + st) * 2 + 1) * 32 + lane) * 16]);
  }

  v8f o[4] = {};
  float mrow[8], lrow[8];
#pragma unroll
  for (int r = 0; r < 8; ++r) { mrow[r] = -3.0e38f; lrow[r] = 0.0f; }

  for (int tc = 0; tc < 32; ++tc) {
    const int t0 = tc * 32;

    // ---- content scores ----
    v8f sc[2] = {};
#pragma unroll
    for (int nsub = 0; nsub < 2; ++nsub) {
#pragma unroll
      for (int d0i = 0; d0i < 2; ++d0i) {
        B16x16 bfr;
        load_frag(bfr, &PK[(((size_t)(bh * 64 + (t0 >> 4) + nsub) * 2 + d0i) * 32 + lane) * 16]);
        sc[nsub] = wmma_bf16(qa[d0i], bfr, sc[nsub]);
      }
    }

    // ---- rel-pos band: 64 aligned table cols covering t-s+1022 for chunk ----
    const int cbase = t0 - s0 - sw0 + 992;       // 16-aligned, may be -16
    const int ct0 = cbase >> 4;
    v8f pc[4] = {};
#pragma unroll
    for (int nt = 0; nt < 4; ++nt) {
      int ct = ct0 + nt;
      ct = ct < 0 ? 0 : (ct > TBL_CT - 1 ? TBL_CT - 1 : ct);
#pragma unroll
      for (int d0i = 0; d0i < 2; ++d0i) {
        B16x16 bfr;
        load_frag(bfr, &PT[(((size_t)d0i * TBL_CT + ct) * 32 + lane) * 16]);
        pc[nt] = wmma_bf16(qa[d0i], bfr, pc[nt]);
      }
    }
#pragma unroll
    for (int nt = 0; nt < 4; ++nt)
#pragma unroll
      for (int r = 0; r < 8; ++r)
        pos_s[w][(r + 8 * half) * 64 + nt * 16 + n15] = f32_to_bf16(pc[nt][r]);
    asm volatile("" ::: "memory");   // keep DS store -> gather order (in-order per wave)

    // ---- gather P[s, t-s+1022] (index always in [15,61]) and add ----
#pragma unroll
    for (int nsub = 0; nsub < 2; ++nsub)
#pragma unroll
      for (int r = 0; r < 8; ++r) {
        const int sg = s0 + sw0 + r + 8 * half;
        const int tg = t0 + nsub * 16 + n15;
        const int li = tg - sg + 1022 - cbase;
        sc[nsub][r] += bf16_to_f32(pos_s[w][(r + 8 * half) * 64 + li]);
      }

    // ---- online softmax over the 32-col chunk ----
    float p0[8], p1[8];
#pragma unroll
    for (int r = 0; r < 8; ++r) {
      float mx = fmaxf(sc[0][r], sc[1][r]);
      mx = fmaxf(mx, __shfl_xor(mx, 1, 32));
      mx = fmaxf(mx, __shfl_xor(mx, 2, 32));
      mx = fmaxf(mx, __shfl_xor(mx, 4, 32));
      mx = fmaxf(mx, __shfl_xor(mx, 8, 32));
      const float mnew  = fmaxf(mrow[r], mx);
      const float alpha = __expf(mrow[r] - mnew);
      p0[r] = __expf(sc[0][r] - mnew);
      p1[r] = __expf(sc[1][r] - mnew);
      float sum = p0[r] + p1[r];
      sum += __shfl_xor(sum, 1, 32);
      sum += __shfl_xor(sum, 2, 32);
      sum += __shfl_xor(sum, 4, 32);
      sum += __shfl_xor(sum, 8, 32);
      lrow[r] = lrow[r] * alpha + sum;
      mrow[r] = mnew;
#pragma unroll
      for (int a = 0; a < 4; ++a) o[a][r] *= alpha;
    }

    // ---- probs C-layout -> A-frag-ready LDS, then O += P x V ----
#pragma unroll
    for (int r = 0; r < 8; ++r) {
      const int m = r + 8 * half;
      const int K0 = n15, K1 = 16 + n15;
      pr_s[w][a_lane(K0, m) * 16 + a_e(K0)] = f32_to_bf16(p0[r]);
      pr_s[w][a_lane(K1, m) * 16 + a_e(K1)] = f32_to_bf16(p1[r]);
    }
    asm volatile("" ::: "memory");
    B16x16 pa;
    load_frag(pa, &pr_s[w][lane * 16]);
#pragma unroll
    for (int a = 0; a < 4; ++a) {
      B16x16 bfr;
      load_frag(bfr, &PV[(((size_t)(bh * 32 + tc) * 4 + a) * 32 + lane) * 16]);
      o[a] = wmma_bf16(pa, bfr, o[a]);
    }
  }

  // Finalize: O /= l, scatter into A-frag-ready PO tiles for the out-proj.
#pragma unroll
  for (int r = 0; r < 8; ++r) {
    const float rl = 1.0f / lrow[r];
    const int m = b * S_LEN + s0 + sw0 + r + 8 * half;   // row = b*S + s
    const int mt = m >> 4, mi = m & 15;
#pragma unroll
    for (int a = 0; a < 4; ++a) {
      const int col = h * 64 + a * 16 + n15;
      const int kt = col >> 5, kin = col & 31;
      PO[(((size_t)(mt * 32 + kt)) * 32 + a_lane(kin, mi)) * 16 + a_e(kin)] =
          f32_to_bf16(o[a][r] * rl);
    }
  }
}

// ---------------------------------------------------------------------------
// Output projection: PO (A-ready) x PWo (B-ready) + bo -> out[S,B,E] f32.
// No LDS: both operands stream as b128 fragment loads from global.
// Each wave: 2 m-tiles x 4 n-tiles; every B-frag feeds 2 wmmas.
// ---------------------------------------------------------------------------
__global__ __launch_bounds__(256) void k_outproj(
    const unsigned short* __restrict__ PO, const unsigned short* __restrict__ PWo,
    const float* __restrict__ bo, float* __restrict__ out) {
  const int m0 = blockIdx.x * 128;
  const int n0 = blockIdx.y * 128;
  const int tid = threadIdx.x;
  const int w = tid >> 5, lane = tid & 31;
  const int half = lane >> 4, n15 = lane & 15;
  const int wm = w & 3, wn = w >> 2;

  v8f acc[2][4] = {};
  const int mt0 = (m0 >> 4) + wm * 2;
  const int ntg0 = (n0 >> 4) + wn * 4;

  for (int kt = 0; kt < 32; ++kt) {
    B16x16 a0, a1;
    load_frag(a0, &PO[((size_t)((mt0 + 0) * 32 + kt) * 32 + lane) * 16]);
    load_frag(a1, &PO[((size_t)((mt0 + 1) * 32 + kt) * 32 + lane) * 16]);
#pragma unroll
    for (int a = 0; a < 4; ++a) {
      B16x16 bfr;
      load_frag(bfr, &PWo[((size_t)(kt * 64 + ntg0 + a) * 32 + lane) * 16]);
      acc[0][a] = wmma_bf16(a0, bfr, acc[0][a]);
      acc[1][a] = wmma_bf16(a1, bfr, acc[1][a]);   // B-frag reused
    }
  }

#pragma unroll
  for (int mi = 0; mi < 2; ++mi) {
#pragma unroll
    for (int a = 0; a < 4; ++a) {
      const int ng = n0 + wn * 64 + a * 16 + n15;
      const float bval = bo[ng];
#pragma unroll
      for (int r = 0; r < 8; ++r) {
        const int mg = m0 + (wm * 2 + mi) * 16 + r + 8 * half;   // row = b*S + s
        const int b = mg >> 10, s = mg & 1023;
        out[((size_t)s * BATCH + b) * EMB + ng] = acc[mi][a][r] + bval;
      }
    }
  }
}

// ---------------------------------------------------------------------------
// Host launch
// ---------------------------------------------------------------------------
extern "C" void kernel_launch(void* const* d_in, const int* in_sizes, int n_in,
                              void* d_out, int out_size, void* d_ws, size_t ws_size,
                              hipStream_t stream) {
  (void)in_sizes; (void)n_in; (void)out_size; (void)ws_size;

  const float* q_in = (const float*)d_in[0];
  const float* k_in = (const float*)d_in[1];
  const float* v_in = (const float*)d_in[2];
  const float* Wq = (const float*)d_in[3];  const float* bq = (const float*)d_in[4];
  const float* Wk = (const float*)d_in[5];  const float* bk = (const float*)d_in[6];
  const float* Wv = (const float*)d_in[7];  const float* bv = (const float*)d_in[8];
  const float* Wo = (const float*)d_in[9];  const float* bo = (const float*)d_in[10];
  const float* tab = (const float*)d_in[11];
  float* out = (float*)d_out;

  char* ws = (char*)d_ws;
  const size_t SZ_W   = (size_t)EMB * EMB * 2;                   // 2 MB each
  const size_t SZ_TT  = (size_t)2 * TBL_CT * 512 * 2;            // 256 KB
  const size_t SZ_HED = (size_t)BATCH * HEADS * S_LEN * HD * 2;  // 8 MB each
  size_t off = 0;
  unsigned short* PWq = (unsigned short*)(ws + off); off += SZ_W;
  unsigned short* PWk = (unsigned short*)(ws + off); off += SZ_W;
  unsigned short* PWv = (unsigned short*)(ws + off); off += SZ_W;
  unsigned short* PWo = (unsigned short*)(ws + off); off += SZ_W;
  unsigned short* PT  = (unsigned short*)(ws + off); off += SZ_TT;
  unsigned short* PQ  = (unsigned short*)(ws + off); off += SZ_HED;
  unsigned short* PK  = (unsigned short*)(ws + off); off += SZ_HED;
  unsigned short* PV  = (unsigned short*)(ws + off); off += SZ_HED;
  unsigned short* PO  = (unsigned short*)(ws + off); off += SZ_HED;

  k_pack_w<<<(EMB * EMB) / 256, 256, 0, stream>>>(Wq, PWq);
  k_pack_w<<<(EMB * EMB) / 256, 256, 0, stream>>>(Wk, PWk);
  k_pack_w<<<(EMB * EMB) / 256, 256, 0, stream>>>(Wv, PWv);
  k_pack_w<<<(EMB * EMB) / 256, 256, 0, stream>>>(Wo, PWo);
  k_pack_tab<<<(2 * TBL_CT * 512 + 255) / 256, 256, 0, stream>>>(tab, PT);

  const float scale = 0.125f;  // hd^-0.5
  k_qkv<<<dim3(MROWS / 128, EMB / 128, 3), 256, 0, stream>>>(
      q_in, k_in, v_in, PWq, PWk, PWv, bq, bk, bv, PQ, PK, PV, scale);

  k_attn<<<dim3(BATCH * HEADS, S_LEN / 64), 128, 0, stream>>>(PQ, PK, PV, PT, PO);

  k_outproj<<<dim3(MROWS / 128, EMB / 128), 256, 0, stream>>>(PO, PWo, bo, out);
}
